// Multi_Head_Masked_Attention_68015102099774
// MI455X (gfx1250) — matrix operand order repaired and checked by reading, not compiled, not run on hardware
//
#include <hip/hip_runtime.h>
#include <hip/hip_bf16.h>

typedef __attribute__((ext_vector_type(16))) __bf16 bf16x16;
typedef __attribute__((ext_vector_type(8)))  __bf16 bf16x8;
typedef __attribute__((ext_vector_type(4)))  __bf16 bf16x4;
typedef __attribute__((ext_vector_type(8)))  float  f32x8;

#define WMMA_BF16(a, b, c) \
  __builtin_amdgcn_wmma_f32_16x16x32_bf16(false, (a), false, (b), (short)0, (c), false, false)

static __device__ inline __bf16 f2bf(float f) { return (__bf16)f; }  // native v_cvt, RNE

static __device__ inline bf16x16 join16(bf16x8 lo, bf16x8 hi) {
  return __builtin_shufflevector(lo, hi, 0,1,2,3,4,5,6,7,8,9,10,11,12,13,14,15);
}

// lane-xor within 16-lane halves via ds_swizzle (and=0x1F, or=0, xor=M)
template<int M>
static __device__ inline float swz(float v) {
  return __builtin_bit_cast(float,
      __builtin_amdgcn_ds_swizzle(__builtin_bit_cast(int, v), 0x1F | (M << 10)));
}
static __device__ inline float rmax16(float v) {
  v = fmaxf(v, swz<1>(v)); v = fmaxf(v, swz<2>(v));
  v = fmaxf(v, swz<4>(v)); v = fmaxf(v, swz<8>(v));
  return v;
}
static __device__ inline float rsum16(float v) {
  v += swz<1>(v); v += swz<2>(v); v += swz<4>(v); v += swz<8>(v);
  return v;
}

// ---------------------------------------------------------------- fp32 -> bf16
__global__ __launch_bounds__(256)
void cvt_bf16_kernel(const float* __restrict__ src, __bf16* __restrict__ dst, int n4) {
  int i = blockIdx.x * 256 + threadIdx.x;
  if (i >= n4) return;
  float4 v = reinterpret_cast<const float4*>(src)[i];
  bf16x4 o;
  o[0] = f2bf(v.x); o[1] = f2bf(v.y); o[2] = f2bf(v.z); o[3] = f2bf(v.w);
  *reinterpret_cast<bf16x4*>(dst + (size_t)i * 4) = o;
}

// ----------------------------------------------------------- WMMA GEMM C=A*W^T
// A: [M,K] bf16 row-major.  W: [N,K] bf16 row-major (torch Linear layout).
// Each wave computes a 32(M) x 64(N) tile.
// MODE 1: bf16 out [M,N];  MODE 2: bf16 out, per-head transposed [B,H,hd,S];
// MODE 3: f32 out [M,N] + bias.
template<int MODE>
__global__ __launch_bounds__(256)
void gemm_wmma_kernel(const __bf16* __restrict__ A, const __bf16* __restrict__ W,
                      void* __restrict__ outp, const float* __restrict__ bias,
                      int M, int N, int K) {
  const int lane = threadIdx.x & 31;
  const int wave = threadIdx.x >> 5;
  const int lc = lane & 15, hi = lane >> 4;
  const int tilesN = N >> 6;
  const int wg = blockIdx.x * 8 + wave;
  const int tm = wg / tilesN;            // 32-row tile
  const int tn = wg % tilesN;            // 64-col tile

  f32x8 acc[2][4] = {};
  const __bf16* arow0 = A + (size_t)(tm * 32 + lc) * K;
  const __bf16* arow1 = arow0 + (size_t)16 * K;
  const __bf16* wr[4];
#pragma unroll
  for (int j = 0; j < 4; ++j) wr[j] = W + (size_t)(tn * 64 + j * 16 + lc) * K;

  for (int k0 = 0; k0 < K; k0 += 32) {
    bf16x16 a0 = join16(*(const bf16x8*)(arow0 + k0 + hi * 16),
                        *(const bf16x8*)(arow0 + k0 + hi * 16 + 8));
    bf16x16 a1 = join16(*(const bf16x8*)(arow1 + k0 + hi * 16),
                        *(const bf16x8*)(arow1 + k0 + hi * 16 + 8));
#pragma unroll
    for (int j = 0; j < 4; ++j) {
      bf16x16 b = *(const bf16x16*)(wr[j] + k0 + hi * 16);
      acc[0][j] = WMMA_BF16(a0, b, acc[0][j]);
      acc[1][j] = WMMA_BF16(a1, b, acc[1][j]);
    }
  }

#pragma unroll
  for (int mi = 0; mi < 2; ++mi)
#pragma unroll
    for (int j = 0; j < 4; ++j)
#pragma unroll
      for (int r = 0; r < 8; ++r) {
        int row = tm * 32 + mi * 16 + hi * 8 + r;
        int col = tn * 64 + j * 16 + lc;
        float v = acc[mi][j][r];
        if (MODE == 1) {
          ((__bf16*)outp)[(size_t)row * N + col] = f2bf(v);
        } else if (MODE == 2) {           // V^T: [B=2, H=16, hd=64, S=2048]
          int bb = row >> 11, s = row & 2047;
          int hh = col >> 6,  dh = col & 63;
          ((__bf16*)outp)[(((size_t)(bb * 16 + hh) * 64 + dh) << 11) + s] = f2bf(v);
        } else {                          // MODE 3
          ((float*)outp)[(size_t)row * N + col] = v + bias[col];
        }
      }
}

// ------------------------------------------------------- flash attention core
// One wave per (b, h, 16-query tile).  Q,K: [B,S,D] bf16.  Vt: [B,H,hd,S] bf16.
// Ctx out: [B,S,D] bf16.  Softmax tracked in exp2 domain (scale folds log2(e)).
__global__ __launch_bounds__(256)
void attn_wmma_kernel(const __bf16* __restrict__ Q, const __bf16* __restrict__ Km,
                      const __bf16* __restrict__ Vt, __bf16* __restrict__ Ctx) {
  __shared__ __align__(16) __bf16 plds[8][16][32];   // per-wave P tile (bf16)
  const int lane = threadIdx.x & 31;
  const int wave = threadIdx.x >> 5;
  const int lc = lane & 15, hi = lane >> 4;
  const int wg = blockIdx.x * 8 + wave;   // 0..4095
  const int b  = wg >> 11;
  const int h  = (wg >> 7) & 15;
  const int qt = wg & 127;
  const float scale2 = 0.125f * 1.44269504088896f;   // (1/sqrt(64)) * log2(e)

  // Q A-fragments (16 rows x 64 dims = 2 chunks of K=32)
  const __bf16* qp = Q + ((size_t)(b * 2048 + qt * 16 + lc)) * 1024 + h * 64;
  bf16x16 qa[2];
#pragma unroll
  for (int ks = 0; ks < 2; ++ks)
    qa[ks] = join16(*(const bf16x8*)(qp + ks * 32 + hi * 16),
                    *(const bf16x8*)(qp + ks * 32 + hi * 16 + 8));

  float mrow[8], lrow[8];
#pragma unroll
  for (int r = 0; r < 8; ++r) { mrow[r] = -1e30f; lrow[r] = 0.0f; }
  f32x8 o[4] = {};

  const __bf16* kp = Km + ((size_t)b * 2048) * 1024 + h * 64;
  const __bf16* vp = Vt + ((size_t)(b * 16 + h) * 64) * 2048;
  const int nkt = (qt >> 1) + 1;          // causal: 32-key tiles covering keys<=q

  auto process_tile = [&](int kb, bool masked, bool prefetch) {
    if (prefetch) {                        // warm next tile's K/V rows in L2/L0
      __builtin_prefetch(kp + (size_t)(kb + 32 + lc) * 1024, 0, 1);
      __builtin_prefetch(vp + (size_t)lc * 2048 + kb + 32, 0, 1);
    }
    // S = Q * K^T for 32 keys -> two 16x16 f32 fragments
    f32x8 s0 = {}, s1 = {};
#pragma unroll
    for (int ks = 0; ks < 2; ++ks) {
      bf16x16 b0 = *(const bf16x16*)(kp + (size_t)(kb + lc) * 1024 + ks * 32 + hi * 16);
      s0 = WMMA_BF16(qa[ks], b0, s0);
      bf16x16 b1 = *(const bf16x16*)(kp + (size_t)(kb + 16 + lc) * 1024 + ks * 32 + hi * 16);
      s1 = WMMA_BF16(qa[ks], b1, s1);
    }
    // online softmax (per row; rows live in 16-lane halves), exp2 domain
#pragma unroll
    for (int r = 0; r < 8; ++r) {
      float v0 = s0[r] * scale2;
      float v1 = s1[r] * scale2;
      if (masked) {                        // only the diagonal tile needs this
        int q_idx = qt * 16 + hi * 8 + r;
        v0 = (kb + lc      <= q_idx) ? v0 : -1e30f;
        v1 = (kb + 16 + lc <= q_idx) ? v1 : -1e30f;
      }
      float mn = fmaxf(mrow[r], rmax16(fmaxf(v0, v1)));
      float p0 = exp2f(v0 - mn);
      float p1 = exp2f(v1 - mn);
      float corr = exp2f(mrow[r] - mn);
      lrow[r] = lrow[r] * corr + rsum16(p0 + p1);
      mrow[r] = mn;
#pragma unroll
      for (int j = 0; j < 4; ++j) o[j][r] *= corr;
      plds[wave][hi * 8 + r][lc]      = f2bf(p0);
      plds[wave][hi * 8 + r][16 + lc] = f2bf(p1);
    }
    asm volatile("s_wait_dscnt 0" ::: "memory");     // C-layout -> A-layout via LDS
    bf16x16 pa = join16(*(const bf16x8*)&plds[wave][lc][hi * 16],
                        *(const bf16x8*)&plds[wave][lc][hi * 16 + 8]);
    // O += P * V   (V^T layout => contiguous B-fragment loads)
#pragma unroll
    for (int j = 0; j < 4; ++j) {
      bf16x16 vb = *(const bf16x16*)(vp + (size_t)(j * 16 + lc) * 2048 + kb + hi * 16);
      o[j] = WMMA_BF16(pa, vb, o[j]);
    }
  };

  for (int kt = 0; kt < nkt - 1; ++kt)      // interior tiles: never masked
    process_tile(kt * 32, false, true);
  process_tile((nkt - 1) * 32, true, false); // diagonal tile: causal mask

#pragma unroll
  for (int r = 0; r < 8; ++r) {
    float inv = 1.0f / lrow[r];
    size_t rowoff = (size_t)(b * 2048 + qt * 16 + hi * 8 + r) * 1024 + h * 64;
#pragma unroll
    for (int j = 0; j < 4; ++j)
      Ctx[rowoff + j * 16 + lc] = f2bf(o[j][r] * inv);
  }
}

// ------------------------------------------------------------------- launcher
extern "C" void kernel_launch(void* const* d_in, const int* in_sizes, int n_in,
                              void* d_out, int out_size, void* d_ws, size_t ws_size,
                              hipStream_t stream) {
  const float* inp = (const float*)d_in[0];   // [2,2048,1024]
  const float* Wq  = (const float*)d_in[1];   // [1024,1024]
  const float* Wk  = (const float*)d_in[2];
  const float* Wv  = (const float*)d_in[3];
  const float* Wo  = (const float*)d_in[4];
  const float* bo  = (const float*)d_in[5];   // [1024]
  float* out = (float*)d_out;

  const size_t NI = (size_t)4096 * 1024;      // B*S*D
  const size_t NW = (size_t)1024 * 1024;
  __bf16* Xb  = (__bf16*)d_ws;
  __bf16* Wqb = Xb  + NI;
  __bf16* Wkb = Wqb + NW;
  __bf16* Wvb = Wkb + NW;
  __bf16* Wob = Wvb + NW;
  __bf16* Qb  = Wob + NW;
  __bf16* Kb  = Qb  + NI;
  __bf16* Vtb = Kb  + NI;
  __bf16* Cb  = Vtb + NI;

  // 1) convert everything to bf16
  cvt_bf16_kernel<<<(int)(NI / 4 / 256), 256, 0, stream>>>(inp, Xb,  (int)(NI / 4));
  cvt_bf16_kernel<<<(int)(NW / 4 / 256), 256, 0, stream>>>(Wq,  Wqb, (int)(NW / 4));
  cvt_bf16_kernel<<<(int)(NW / 4 / 256), 256, 0, stream>>>(Wk,  Wkb, (int)(NW / 4));
  cvt_bf16_kernel<<<(int)(NW / 4 / 256), 256, 0, stream>>>(Wv,  Wvb, (int)(NW / 4));
  cvt_bf16_kernel<<<(int)(NW / 4 / 256), 256, 0, stream>>>(Wo,  Wob, (int)(NW / 4));

  // 2) Q/K/V projections: M=4096, N=1024, K=1024; 2048 waves -> 256 blocks
  const int gemmBlocks = (4096 / 32) * (1024 / 64) / 8;   // 256
  gemm_wmma_kernel<1><<<gemmBlocks, 256, 0, stream>>>(Xb, Wqb, Qb,  nullptr, 4096, 1024, 1024);
  gemm_wmma_kernel<1><<<gemmBlocks, 256, 0, stream>>>(Xb, Wkb, Kb,  nullptr, 4096, 1024, 1024);
  gemm_wmma_kernel<2><<<gemmBlocks, 256, 0, stream>>>(Xb, Wvb, Vtb, nullptr, 4096, 1024, 1024);

  // 3) causal flash attention: 4096 waves -> 512 blocks
  attn_wmma_kernel<<<512, 256, 0, stream>>>(Qb, Kb, Vtb, Cb);

  // 4) output projection + bias, f32 out
  gemm_wmma_kernel<3><<<gemmBlocks, 256, 0, stream>>>(Cb, Wob, out, bo, 4096, 1024, 1024);
}
